// ViTBackbone_46892452938338
// MI455X (gfx1250) — matrix operand order repaired and checked by reading, not compiled, not run on hardware
//
#include <hip/hip_runtime.h>
#include <math.h>

// ---------------- model constants ----------------
#define BATCH 32
#define DMODEL 768
#define NHEAD 12
#define NLAYER 12
#define DHEAD 64
#define LFULL 197
#define NPATCH 196
#define KTOP 100
#define LSEL 101   // 1 + KTOP
#define HP 14
#define PSZ 16

typedef __attribute__((ext_vector_type(8)))  float    v8f;
typedef __attribute__((ext_vector_type(16))) _Float16 v16h;

union FragU { v16h h; unsigned u[8]; };
union PackU { _Float16 h[16]; uint4 v4[2]; };

static __device__ __forceinline__ int imin(int a, int b) { return a < b ? a : b; }

// =====================================================================
// Generic WMMA GEMM:  C[b,h] = epilogue( scale * A[b,h] x B[b,h] )
//   A[m][k] = A + b*sAb + h*sAh + m*sAm + k*sAk
//   B[k][n] = B + b*sBb + h*sBh + k*sBk + n*sBn
//   C[m][n] = C + b*sCb + h*sCh + m*ldc + n    (residual uses same index)
// 128 threads (4 waves, wave32); 128x64 tile; K stepped by 32.
// Each wave owns 32 rows x 64 cols = 8 WMMA tiles (2 A-frags x 4 B-frags).
// Staging is software-pipelined: next k-tile is loaded into registers
// while the current tile's v_wmma_f32_16x16x32_f16 ops execute.
// =====================================================================
__global__ __launch_bounds__(128)
void gemm_wmma_kernel(const float* __restrict__ A, const float* __restrict__ B,
                      float* __restrict__ C,
                      const float* __restrict__ bias, const float* __restrict__ residual,
                      int M, int N, int K,
                      long sAm, long sAk, long sAb, long sAh,
                      long sBk, long sBn, long sBb, long sBh,
                      long ldc, long sCb, long sCh,
                      int Hdim, float scale, int do_gelu)
{
    __shared__ _Float16 sA[128 * 32];  // [m][k]
    __shared__ _Float16 sB[64 * 32];   // [n][k]  (transposed stage)

    const int batch = blockIdx.z;
    const int bb = batch / Hdim;
    const int hh = batch - bb * Hdim;
    const float* Ap = A + bb * sAb + hh * sAh;
    const float* Bp = B + bb * sBb + hh * sBh;
    const long   cbase = (long)bb * sCb + (long)hh * sCh;

    const int m0 = blockIdx.y * 128;
    const int n0 = blockIdx.x * 64;

    const int tid  = threadIdx.x;
    const int lane = tid & 31;
    const int wv   = tid >> 5;        // wave 0..3 -> row group within each 64-row half
    const int lh   = lane & 15;
    const int hi   = lane >> 4;       // 0 or 1

    // ---- staging ownership ----
    // B: 1 row x 16 contiguous k per thread (64x32 tile)
    // A: 2 rows (r, r+64) x 16 contiguous k per thread (128x32 tile)
    const int srow  = tid >> 1;           // 0..63
    const int skoff = (tid & 1) * 16;     // 0 or 16
    const bool mok0 = (m0 + srow) < M;
    const bool mok1 = (m0 + 64 + srow) < M;
    const bool nok  = (n0 + srow) < N;
    const int  mc0  = imin(m0 + srow, M - 1);
    const int  mc1  = imin(m0 + 64 + srow, M - 1);
    const int  nc   = imin(n0 + srow, N - 1);
    const float* pa0 = Ap + (long)mc0 * sAm;   // + k*sAk per element
    const float* pa1 = Ap + (long)mc1 * sAm;
    const float* pb  = Bp + (long)nc  * sBn;   // + k*sBk per element
    // uniform: whole 128x64 tile in-bounds in m and n?
    const bool edge_mn = (m0 + 128 > M) || (n0 + 64 > N);

    v8f acc[2][4];
    {
        v8f z = {0.f,0.f,0.f,0.f,0.f,0.f,0.f,0.f};
        #pragma unroll
        for (int mi = 0; mi < 2; ++mi)
            #pragma unroll
            for (int nt = 0; nt < 4; ++nt) acc[mi][nt] = z;
    }

    float ra0[16], ra1[16], rb[16];

    // ---- prologue: load first k-tile into registers ----
    {
        const int kb = skoff;                       // k0 == 0
        if (!edge_mn && 32 <= K) {
            #pragma unroll
            for (int j = 0; j < 16; ++j) {
                ra0[j] = pa0[(long)(kb + j) * sAk];
                ra1[j] = pa1[(long)(kb + j) * sAk];
                rb[j]  = pb [(long)(kb + j) * sBk];
            }
        } else {
            #pragma unroll
            for (int j = 0; j < 16; ++j) {
                int kc = imin(kb + j, K - 1);
                float a0 = pa0[(long)kc * sAk];
                float a1 = pa1[(long)kc * sAk];
                float bv = pb [(long)kc * sBk];
                bool kok = (kb + j) < K;
                ra0[j] = (mok0 && kok) ? a0 : 0.f;
                ra1[j] = (mok1 && kok) ? a1 : 0.f;
                rb[j]  = (nok  && kok) ? bv : 0.f;
            }
        }
    }

    for (int k0 = 0; k0 < K; k0 += 32) {
        // ---- commit registers to LDS (packed 128-bit stores) ----
        __syncthreads();   // previous iteration's fragment reads are done
        {
            PackU pk0, pk1, pkb;
            #pragma unroll
            for (int j = 0; j < 16; ++j) {
                pk0.h[j] = (_Float16)ra0[j];
                pk1.h[j] = (_Float16)ra1[j];
                pkb.h[j] = (_Float16)rb[j];
            }
            uint4* d0 = (uint4*)&sA[srow * 32 + skoff];
            uint4* d1 = (uint4*)&sA[(64 + srow) * 32 + skoff];
            uint4* db = (uint4*)&sB[srow * 32 + skoff];
            d0[0] = pk0.v4[0]; d0[1] = pk0.v4[1];
            d1[0] = pk1.v4[0]; d1[1] = pk1.v4[1];
            db[0] = pkb.v4[0]; db[1] = pkb.v4[1];
        }
        __syncthreads();

        // ---- prefetch next k-tile into registers (overlaps with WMMA) ----
        const int kn = k0 + 32;
        if (kn < K) {
            const int kb = kn + skoff;
            if (!edge_mn && kn + 32 <= K) {
                #pragma unroll
                for (int j = 0; j < 16; ++j) {
                    ra0[j] = pa0[(long)(kb + j) * sAk];
                    ra1[j] = pa1[(long)(kb + j) * sAk];
                    rb[j]  = pb [(long)(kb + j) * sBk];
                }
            } else {
                #pragma unroll
                for (int j = 0; j < 16; ++j) {
                    int kc = imin(kb + j, K - 1);
                    float a0 = pa0[(long)kc * sAk];
                    float a1 = pa1[(long)kc * sAk];
                    float bv = pb [(long)kc * sBk];
                    bool kok = (kb + j) < K;
                    ra0[j] = (mok0 && kok) ? a0 : 0.f;
                    ra1[j] = (mok1 && kok) ? a1 : 0.f;
                    rb[j]  = (nok  && kok) ? bv : 0.f;
                }
            }
        }

        // ---- preload all fragments, then issue 8 WMMAs ----
        FragU af[2], bf[4];
        #pragma unroll
        for (int mi = 0; mi < 2; ++mi) {
            const int arow = (mi * 64 + wv * 16 + lh) * 32;
            #pragma unroll
            for (int j = 0; j < 8; ++j) {
                int kk = (j < 4) ? (2 * j + hi * 8) : (16 + 2 * (j - 4) + hi * 8);
                af[mi].u[j] = *(const unsigned*)&sA[arow + kk];
            }
        }
        #pragma unroll
        for (int nt = 0; nt < 4; ++nt) {
            const int brow = (nt * 16 + lh) * 32;
            #pragma unroll
            for (int j = 0; j < 8; ++j) {
                int kk = (j < 4) ? (2 * j + hi * 8) : (16 + 2 * (j - 4) + hi * 8);
                bf[nt].u[j] = *(const unsigned*)&sB[brow + kk];
            }
        }
        #pragma unroll
        for (int nt = 0; nt < 4; ++nt) {
            acc[0][nt] = __builtin_amdgcn_wmma_f32_16x16x32_f16(
                false, af[0].h, false, bf[nt].h, (short)0, acc[0][nt], false, false);
            acc[1][nt] = __builtin_amdgcn_wmma_f32_16x16x32_f16(
                false, af[1].h, false, bf[nt].h, (short)0, acc[1][nt], false, false);
        }
    }

    // ---- epilogue + store ----
    #pragma unroll
    for (int mi = 0; mi < 2; ++mi) {
        #pragma unroll
        for (int nt = 0; nt < 4; ++nt) {
            int n = n0 + nt * 16 + lh;
            #pragma unroll
            for (int r = 0; r < 8; ++r) {
                int m = m0 + mi * 64 + wv * 16 + r + hi * 8;
                if (m < M && n < N) {
                    float v = acc[mi][nt][r] * scale;
                    if (bias) v += bias[n];
                    if (do_gelu) v = 0.5f * v * (1.f + erff(v * 0.70710678118654752f));
                    long ci = cbase + (long)m * ldc + n;
                    if (residual) v += residual[ci];
                    C[ci] = v;
                }
            }
        }
    }
}

// ---------------- LayerNorm over last dim 768, one block per row ----------------
__global__ __launch_bounds__(256)
void ln_kernel(const float* __restrict__ x, float* __restrict__ y,
               const float* __restrict__ s, const float* __restrict__ b)
{
    const int row = blockIdx.x;
    const int tid = threadIdx.x;
    const float* xr = x + (long)row * DMODEL;
    float v0 = xr[tid], v1 = xr[tid + 256], v2 = xr[tid + 512];
    __shared__ float red[256];
    red[tid] = v0 + v1 + v2;
    __syncthreads();
    for (int st = 128; st > 0; st >>= 1) { if (tid < st) red[tid] += red[tid + st]; __syncthreads(); }
    float mean = red[0] * (1.f / DMODEL);
    __syncthreads();
    float d0 = v0 - mean, d1 = v1 - mean, d2 = v2 - mean;
    red[tid] = d0 * d0 + d1 * d1 + d2 * d2;
    __syncthreads();
    for (int st = 128; st > 0; st >>= 1) { if (tid < st) red[tid] += red[tid + st]; __syncthreads(); }
    float inv = rsqrtf(red[0] * (1.f / DMODEL) + 1e-6f);
    float* yr = y + (long)row * DMODEL;
    yr[tid]       = d0 * inv * s[tid]       + b[tid];
    yr[tid + 256] = d1 * inv * s[tid + 256] + b[tid + 256];
    yr[tid + 512] = d2 * inv * s[tid + 512] + b[tid + 512];
}

// ---------------- softmax over rows of length L (in place) ----------------
__global__ __launch_bounds__(128)
void softmax_kernel(float* __restrict__ x, int L)
{
    float* p = x + (long)blockIdx.x * L;
    const int tid = threadIdx.x;
    __shared__ float red[128];
    float m = -3.4e38f;
    for (int l = tid; l < L; l += 128) m = fmaxf(m, p[l]);
    red[tid] = m; __syncthreads();
    for (int st = 64; st > 0; st >>= 1) { if (tid < st) red[tid] = fmaxf(red[tid], red[tid + st]); __syncthreads(); }
    m = red[0]; __syncthreads();
    float sum = 0.f;
    for (int l = tid; l < L; l += 128) { float e = __expf(p[l] - m); p[l] = e; sum += e; }
    red[tid] = sum; __syncthreads();
    for (int st = 64; st > 0; st >>= 1) { if (tid < st) red[tid] += red[tid + st]; __syncthreads(); }
    float inv = 1.f / red[0];
    for (int l = tid; l < L; l += 128) p[l] *= inv;
}

// ---------------- patchify: x[B,3,224,224] -> p[B*196, 768] ----------------
__global__ void im2row_kernel(const float* __restrict__ x, float* __restrict__ p)
{
    long e = (long)blockIdx.x * 256 + threadIdx.x;
    if (e >= (long)BATCH * NPATCH * DMODEL) return;
    int kk = e % DMODEL; long r = e / DMODEL;
    int pidx = r % NPATCH; int b = r / NPATCH;
    int c = kk >> 8, py = (kk >> 4) & 15, px = kk & 15;
    int ph = pidx / HP, pw = pidx % HP;
    p[e] = x[(((long)(b * 3 + c) * 224) + ph * PSZ + py) * 224 + pw * PSZ + px];
}

// ---------------- embed: t = [cls ; patch_out] + pos ----------------
__global__ void embed_kernel(const float* __restrict__ po, const float* __restrict__ cls,
                             const float* __restrict__ pos, float* __restrict__ t)
{
    long e = (long)blockIdx.x * 256 + threadIdx.x;
    if (e >= (long)BATCH * LFULL * DMODEL) return;
    int d = e % DMODEL; long r = e / DMODEL;
    int tok = r % LFULL; int b = r / LFULL;
    float v = (tok == 0) ? cls[d] : po[((long)(b * NPATCH + tok - 1)) * DMODEL + d];
    t[e] = v + pos[(long)tok * DMODEL + d];
}

// ---------------- layer-10 re-attention: query-row-0 scores ----------------
__global__ __launch_bounds__(256)
void score0_kernel(const float* __restrict__ qkv2, float* __restrict__ sc0)
{
    int bh = blockIdx.x;             // b*12 + h
    int b = bh / NHEAD, h = bh % NHEAD;
    const float* q0 = qkv2 + ((long)b * LFULL) * 1536 + h * DHEAD;
    for (int l = threadIdx.x; l < LFULL; l += 256) {
        const float* kr = qkv2 + ((long)(b * LFULL + l)) * 1536 + 768 + h * DHEAD;
        float s = 0.f;
        #pragma unroll 8
        for (int d = 0; d < DHEAD; ++d) s += q0[d] * kr[d];
        sc0[(long)bh * LFULL + l] = s * 0.125f;
    }
}

// ---------------- A_hat[b,p] = sum_h w0[b,h,p+1] ----------------
__global__ void ahat_kernel(const float* __restrict__ w0, float* __restrict__ ah)
{
    int e = blockIdx.x * 256 + threadIdx.x;
    if (e >= BATCH * NPATCH) return;
    int b = e / NPATCH, p = e % NPATCH;
    float s = 0.f;
    for (int h = 0; h < NHEAD; ++h) s += w0[((long)(b * NHEAD + h)) * LFULL + p + 1];
    ah[e] = s;
}

// ---------------- O = A_hat * (1 + sigmoid(t[:,1:] @ imp_w + imp_b)) ----------------
__global__ __launch_bounds__(128)
void impO_kernel(const float* __restrict__ t, const float* __restrict__ iw,
                 const float* __restrict__ ib, const float* __restrict__ ah,
                 float* __restrict__ O)
{
    int bp = blockIdx.x;                       // b*196 + p
    int b = bp / NPATCH, p = bp % NPATCH;
    const float* xr = t + ((long)(b * LFULL + p + 1)) * DMODEL;
    int tid = threadIdx.x;
    float s = 0.f;
    for (int d = tid; d < DMODEL; d += 128) s += xr[d] * iw[d];
    __shared__ float red[128];
    red[tid] = s; __syncthreads();
    for (int st = 64; st > 0; st >>= 1) { if (tid < st) red[tid] += red[tid + st]; __syncthreads(); }
    if (tid == 0) {
        float z = 1.f / (1.f + __expf(-(red[0] + ib[0])));
        O[bp] = ah[bp] * (1.f + z);
    }
}

// ---------------- top-k (k=100) per batch, descending, low-index tie-break ----------------
__global__ __launch_bounds__(256)
void topk_kernel(const float* __restrict__ O, int* __restrict__ idx)
{
    __shared__ float vals[NPATCH];
    __shared__ float rv[256];
    __shared__ int   ri[256];
    const int b = blockIdx.x, tid = threadIdx.x;
    if (tid < NPATCH) vals[tid] = O[(long)b * NPATCH + tid];
    __syncthreads();
    for (int it = 0; it < KTOP; ++it) {
        float bv = -3.4e38f; int bi = -1;
        for (int j = tid; j < NPATCH; j += 256) {
            float v = vals[j];
            if (v > bv || (v == bv && (unsigned)j < (unsigned)bi)) { bv = v; bi = j; }
        }
        rv[tid] = bv; ri[tid] = bi; __syncthreads();
        for (int st = 128; st > 0; st >>= 1) {
            if (tid < st) {
                if (rv[tid + st] > rv[tid] ||
                    (rv[tid + st] == rv[tid] && (unsigned)ri[tid + st] < (unsigned)ri[tid])) {
                    rv[tid] = rv[tid + st]; ri[tid] = ri[tid + st];
                }
            }
            __syncthreads();
        }
        if (tid == 0) { idx[b * KTOP + it] = ri[0]; vals[ri[0]] = -3.4e38f; }
        __syncthreads();
    }
}

// ---------------- gather selected tokens; emit inp and top_patches output ----------------
__global__ void build_inp_kernel(const float* __restrict__ t, const float* __restrict__ pos,
                                 const int* __restrict__ idx, float* __restrict__ inp,
                                 float* __restrict__ top_out)
{
    long e = (long)blockIdx.x * 256 + threadIdx.x;
    if (e >= (long)BATCH * LSEL * DMODEL) return;
    int d = e % DMODEL; long r = e / DMODEL;
    int tok = r % LSEL; int b = r / LSEL;
    if (tok == 0) {
        inp[e] = t[((long)b * LFULL) * DMODEL + d] + pos[d];
    } else {
        int st = idx[b * KTOP + tok - 1] + 1;
        float tv = t[((long)(b * LFULL + st)) * DMODEL + d];
        inp[e] = tv + pos[(long)st * DMODEL + d];
        top_out[((long)(b * KTOP + tok - 1)) * DMODEL + d] = tv;
    }
}

// =====================================================================
// host side
// =====================================================================
static void launch_gemm(hipStream_t st, const float* A, const float* B, float* C,
                        const float* bias, const float* res,
                        int M, int N, int K,
                        long sAm, long sAk, long sAb, long sAh,
                        long sBk, long sBn, long sBb, long sBh,
                        long ldc, long sCb, long sCh,
                        int Hdim, int nbatch, float scale, int gelu)
{
    dim3 g((N + 63) / 64, (M + 127) / 128, nbatch);
    gemm_wmma_kernel<<<g, 128, 0, st>>>(A, B, C, bias, res, M, N, K,
                                        sAm, sAk, sAb, sAh, sBk, sBn, sBb, sBh,
                                        ldc, sCb, sCh, Hdim, scale, gelu);
}

struct Weights {
    const float *ln1s, *ln1b, *qkvw, *qkvb, *projw, *projb;
    const float *ln2s, *ln2b, *w1, *b1, *w2, *b2;
};

// one transformer block, in place on x[B, L, 768]
static void run_block(hipStream_t st, float* x, int L, int i, const Weights& W,
                      float* XLN, float* QKV, float* SCR, float* ATT, float* MLPH)
{
    const int rows = BATCH * L;
    ln_kernel<<<rows, 256, 0, st>>>(x, XLN, W.ln1s + i * DMODEL, W.ln1b + i * DMODEL);
    // qkv = ln(x) @ Wqkv + b : [rows,2304]
    launch_gemm(st, XLN, W.qkvw + (long)i * DMODEL * 3 * DMODEL, QKV,
                W.qkvb + i * 3 * DMODEL, nullptr,
                rows, 3 * DMODEL, DMODEL,
                DMODEL, 1, 0, 0,  3 * DMODEL, 1, 0, 0,
                3 * DMODEL, 0, 0, 1, 1, 1.f, 0);
    // scores[b,h] = q @ k^T * 1/8 : batched (32*12), M=N=L, K=64
    launch_gemm(st, QKV, QKV + 768, SCR, nullptr, nullptr,
                L, L, DHEAD,
                3 * DMODEL, 1, (long)L * 3 * DMODEL, DHEAD,
                1, 3 * DMODEL, (long)L * 3 * DMODEL, DHEAD,
                L, (long)NHEAD * L * L, (long)L * L,
                NHEAD, BATCH * NHEAD, 0.125f, 0);
    softmax_kernel<<<BATCH * NHEAD * L, 128, 0, st>>>(SCR, L);
    // attn_out[b, :, h*64:...] = w @ v : batched, M=L, N=64, K=L
    launch_gemm(st, SCR, QKV + 1536, ATT, nullptr, nullptr,
                L, DHEAD, L,
                L, 1, (long)NHEAD * L * L, (long)L * L,
                3 * DMODEL, 1, (long)L * 3 * DMODEL, DHEAD,
                DMODEL, (long)L * DMODEL, DHEAD,
                NHEAD, BATCH * NHEAD, 1.f, 0);
    // x += attn_out @ Wproj + b
    launch_gemm(st, ATT, W.projw + (long)i * DMODEL * DMODEL, x,
                W.projb + i * DMODEL, x,
                rows, DMODEL, DMODEL,
                DMODEL, 1, 0, 0,  DMODEL, 1, 0, 0,
                DMODEL, 0, 0, 1, 1, 1.f, 0);
    ln_kernel<<<rows, 256, 0, st>>>(x, XLN, W.ln2s + i * DMODEL, W.ln2b + i * DMODEL);
    // h = gelu(ln(x) @ W1 + b1)
    launch_gemm(st, XLN, W.w1 + (long)i * DMODEL * 4 * DMODEL, MLPH,
                W.b1 + i * 4 * DMODEL, nullptr,
                rows, 4 * DMODEL, DMODEL,
                DMODEL, 1, 0, 0,  4 * DMODEL, 1, 0, 0,
                4 * DMODEL, 0, 0, 1, 1, 1.f, 1);
    // x += h @ W2 + b2
    launch_gemm(st, MLPH, W.w2 + (long)i * 4 * DMODEL * DMODEL, x,
                W.b2 + i * DMODEL, x,
                rows, DMODEL, 4 * DMODEL,
                4 * DMODEL, 1, 0, 0,  DMODEL, 1, 0, 0,
                DMODEL, 0, 0, 1, 1, 1.f, 0);
}

extern "C" void kernel_launch(void* const* d_in, const int* in_sizes, int n_in,
                              void* d_out, int out_size, void* d_ws, size_t ws_size,
                              hipStream_t stream)
{
    const float* X      = (const float*)d_in[0];
    const float* conv_w = (const float*)d_in[1];
    const float* conv_b = (const float*)d_in[2];
    const float* cls    = (const float*)d_in[3];
    const float* pos    = (const float*)d_in[4];
    Weights W;
    W.ln1s  = (const float*)d_in[5];
    W.ln1b  = (const float*)d_in[6];
    W.qkvw  = (const float*)d_in[7];
    W.qkvb  = (const float*)d_in[8];
    W.projw = (const float*)d_in[9];
    W.projb = (const float*)d_in[10];
    W.ln2s  = (const float*)d_in[11];
    W.ln2b  = (const float*)d_in[12];
    W.w1    = (const float*)d_in[13];
    W.b1    = (const float*)d_in[14];
    W.w2    = (const float*)d_in[15];
    W.b2    = (const float*)d_in[16];
    const float* flns = (const float*)d_in[17];
    const float* flnb = (const float*)d_in[18];
    const float* impw = (const float*)d_in[19];
    const float* impb = (const float*)d_in[20];

    float* ws = (float*)d_ws;
    long off = 0;
    float* T    = ws + off; off += (long)BATCH * LFULL * DMODEL;
    float* XLN  = ws + off; off += (long)BATCH * LFULL * DMODEL;
    float* QKV  = ws + off; off += (long)BATCH * LFULL * 3 * DMODEL;
    float* SCR  = ws + off; off += (long)BATCH * NHEAD * LFULL * LFULL;
    float* ATT  = ws + off; off += (long)BATCH * LFULL * DMODEL;
    float* MLPH = ws + off; off += (long)BATCH * LFULL * 4 * DMODEL;
    float* PBUF = ws + off; off += (long)BATCH * NPATCH * DMODEL;
    float* POUT = ws + off; off += (long)BATCH * NPATCH * DMODEL;
    float* SC0  = ws + off; off += (long)BATCH * NHEAD * LFULL;
    float* AH   = ws + off; off += (long)BATCH * NPATCH;
    float* OV   = ws + off; off += (long)BATCH * NPATCH;
    float* INP  = ws + off; off += (long)BATCH * LSEL * DMODEL;
    int*   IDX  = (int*)(ws + off); off += (long)BATCH * KTOP;

    float* out_main    = (float*)d_out;                                  // [32,101,768]
    float* out_patches = (float*)d_out + (long)BATCH * LSEL * DMODEL;    // [32,100,768]

    // ---- patch embedding ----
    {
        long n = (long)BATCH * NPATCH * DMODEL;
        im2row_kernel<<<(unsigned)((n + 255) / 256), 256, 0, stream>>>(X, PBUF);
        // p @ conv_w^T + conv_b : B[k][d] = conv_w[d*768 + k]
        launch_gemm(stream, PBUF, conv_w, POUT, conv_b, nullptr,
                    BATCH * NPATCH, DMODEL, DMODEL,
                    DMODEL, 1, 0, 0,  1, DMODEL, 0, 0,
                    DMODEL, 0, 0, 1, 1, 1.f, 0);
        long m = (long)BATCH * LFULL * DMODEL;
        embed_kernel<<<(unsigned)((m + 255) / 256), 256, 0, stream>>>(POUT, cls, pos, T);
    }

    // ---- blocks 0..10 ----
    for (int i = 0; i < NLAYER - 1; ++i)
        run_block(stream, T, LFULL, i, W, XLN, QKV, SCR, ATT, MLPH);

    // ---- layer-10 re-attention (no LN), only q/k columns needed ----
    launch_gemm(stream, T, W.qkvw + (long)(NLAYER - 2) * DMODEL * 3 * DMODEL, QKV,
                W.qkvb + (NLAYER - 2) * 3 * DMODEL, nullptr,
                BATCH * LFULL, 2 * DMODEL, DMODEL,
                DMODEL, 1, 0, 0,  3 * DMODEL, 1, 0, 0,
                2 * DMODEL, 0, 0, 1, 1, 1.f, 0);
    score0_kernel<<<BATCH * NHEAD, 256, 0, stream>>>(QKV, SC0);
    softmax_kernel<<<BATCH * NHEAD, 128, 0, stream>>>(SC0, LFULL);
    ahat_kernel<<<(BATCH * NPATCH + 255) / 256, 256, 0, stream>>>(SC0, AH);

    // ---- importance, O, top-k, gather ----
    impO_kernel<<<BATCH * NPATCH, 128, 0, stream>>>(T, impw, impb, AH, OV);
    topk_kernel<<<BATCH, 256, 0, stream>>>(OV, IDX);
    {
        long n = (long)BATCH * LSEL * DMODEL;
        build_inp_kernel<<<(unsigned)((n + 255) / 256), 256, 0, stream>>>(T, pos, IDX, INP, out_patches);
    }

    // ---- final block (layer 11) on selected tokens, then final LN -> d_out ----
    run_block(stream, INP, LSEL, NLAYER - 1, W, XLN, QKV, SCR, ATT, MLPH);
    ln_kernel<<<BATCH * LSEL, 256, 0, stream>>>(INP, out_main, flns, flnb);
}
